// CrossModalAttention_6253472383370
// MI455X (gfx1250) — matrix operand order repaired and checked by reading, not compiled, not run on hardware
//
#include <hip/hip_runtime.h>
#include <hip/hip_bf16.h>

typedef __attribute__((ext_vector_type(16))) _Float16 v16h;
typedef __attribute__((ext_vector_type(8)))  _Float16 h8;
typedef __attribute__((ext_vector_type(8)))  float    v8f;

#define WMMA_F16(acc, a, b) \
  __builtin_amdgcn_wmma_f32_16x16x32_f16(false, (a), false, (b), (short)0, (acc), false, false)

#if __has_builtin(__builtin_amdgcn_global_load_async_to_lds_b128)
#define HAVE_ASYNC_LDS 1
typedef int i128v __attribute__((vector_size(16)));            // native clang vector, not HIP_vector_type
typedef __attribute__((address_space(1))) i128v gas_i128;      // global (prints as __device__)
typedef __attribute__((address_space(3))) i128v las_i128;      // LDS    (prints as __shared__)
#endif

constexpr int Bc  = 16;
constexpr int Sc  = 512;
constexpr int VDc = 768;
constexpr int Hc  = 1024;
constexpr int NHc = 16;
constexpr int HDc = 64;
constexpr int BSc = Bc * Sc;            // 8192 rows of text
constexpr float SCALEc = 0.125f;        // HD^-0.5

// ---------------------------------------------------------------------------
// 1) abs-mean reduction (for BitNet scale s = mean|W|)
// ---------------------------------------------------------------------------
__global__ void absmean_kernel(const float* __restrict__ W, float* sums, int slot, int n) {
    __shared__ float red[256];
    float s = 0.f;
    for (int i = blockIdx.x * blockDim.x + threadIdx.x; i < n; i += gridDim.x * blockDim.x)
        s += fabsf(W[i]);
    red[threadIdx.x] = s;
    __syncthreads();
    for (int st = 128; st > 0; st >>= 1) {
        if (threadIdx.x < st) red[threadIdx.x] += red[threadIdx.x + st];
        __syncthreads();
    }
    if (threadIdx.x == 0) atomicAdd(&sums[slot], red[0]);
}

// ---------------------------------------------------------------------------
// 2) ternary-quantize W [N,K] and pack into the WMMA B-fragment lane layout:
//      Bp[((kt*tilesN + tn)*512) + lane*16 + e] = q(W[n][k]) * s
//      n = tn*16 + (lane&15);  k = kt*32 + (lane>>4)*16 + e
// ---------------------------------------------------------------------------
__global__ void quant_pack_kernel(const float* __restrict__ W, const float* __restrict__ sums,
                                  int slot, _Float16* __restrict__ Bp, int K, int N,
                                  float invCount) {
    int t = blockIdx.x * blockDim.x + threadIdx.x;
    if (t >= K * N) return;
    float s = sums[slot] * invCount;
    s = fminf(fmaxf(s, 1e-5f), 1000.f);
    int e    = t & 15;
    int lane = (t >> 4) & 31;
    int tile = t >> 9;
    int tilesN = N >> 4;
    int tn = tile % tilesN;
    int kt = tile / tilesN;
    int n = tn * 16 + (lane & 15);
    int k = kt * 32 + (lane >> 4) * 16 + e;
    float w  = W[(size_t)n * K + k];
    float wn = fminf(fmaxf(w / s, -10.f), 10.f);
    float q  = (wn > 0.66666667f) ? 1.f : ((wn < -0.66666667f) ? -1.f : 0.f);
    Bp[t] = (_Float16)(q * s);
}

// ---------------------------------------------------------------------------
// 3) f32 -> f16 conversion
// ---------------------------------------------------------------------------
__global__ void cvt_h_kernel(const float* __restrict__ x, _Float16* __restrict__ y, int n) {
    for (int i = blockIdx.x * blockDim.x + threadIdx.x; i < n; i += gridDim.x * blockDim.x)
        y[i] = (_Float16)x[i];
}

// ---------------------------------------------------------------------------
// A-fragment builder (contiguous k = k0 + 16*half + s, same order as packed B): two contiguous b128 loads
// ---------------------------------------------------------------------------
__device__ __forceinline__ v16h afrag(const _Float16* __restrict__ arow, int k0, int half) {
    h8 x0 = *(const h8*)(arow + k0 + half * 16);
    h8 x1 = *(const h8*)(arow + k0 + half * 16 + 8);
    return __builtin_shufflevector(x0, x1, 0, 1, 2, 3, 4, 5, 6, 7,
                                   8, 9, 10, 11, 12, 13, 14, 15);
}

// ---------------------------------------------------------------------------
// Small GEMM (M == 16): one wave per 16x16 tile. C = A*Bq^T + bias.
// ---------------------------------------------------------------------------
__global__ void gemm_bias_kernel(const _Float16* __restrict__ A, const _Float16* __restrict__ Bp,
                                 const float* __restrict__ bias, float* __restrict__ C,
                                 int M, int N, int K) {
    int lane = threadIdx.x & 31;
    int wid  = threadIdx.x >> 5;
    int tilesN = N >> 4;
    int tile = blockIdx.x * (blockDim.x >> 5) + wid;
    int totalTiles = (M >> 4) * tilesN;
    if (tile >= totalTiles) return;
    int tm = tile / tilesN, tn = tile - tm * tilesN;
    int l16 = lane & 15, half = lane >> 4;
    int col = tn * 16 + l16;
    v8f acc;
#pragma unroll
    for (int i = 0; i < 8; i++) acc[i] = bias[col];
    const _Float16* arow = A + (size_t)(tm * 16 + l16) * K;
    const _Float16* bp = Bp + (size_t)tn * 512 + lane * 16;
    size_t bstride = (size_t)tilesN * 512;
    for (int k0 = 0; k0 < K; k0 += 32, bp += bstride) {
        v16h a = afrag(arow, k0, half);
        v16h b = *(const v16h*)bp;
        __builtin_prefetch(bp + bstride, 0, 1);
        acc = WMMA_F16(acc, a, b);
    }
#pragma unroll
    for (int r = 0; r < 8; r++)
        C[(size_t)(tm * 16 + r + 8 * half) * N + col] = acc[r];
}

// ---------------------------------------------------------------------------
// Big GEMM: each wave computes 4 M-tiles x 1 N-tile (B fragment reused 4x).
// ---------------------------------------------------------------------------
__global__ void gemm_bias4_kernel(const _Float16* __restrict__ A, const _Float16* __restrict__ Bp,
                                  const float* __restrict__ bias, float* __restrict__ C,
                                  int M, int N, int K) {
    int lane = threadIdx.x & 31;
    int wid  = threadIdx.x >> 5;
    int tilesN = N >> 4;
    int tilesM4 = M >> 6;
    int tile = blockIdx.x * (blockDim.x >> 5) + wid;
    if (tile >= tilesM4 * tilesN) return;
    int tm4 = tile / tilesN, tn = tile - tm4 * tilesN;
    int l16 = lane & 15, half = lane >> 4;
    int col = tn * 16 + l16;
    float bv = bias[col];
    v8f acc0, acc1, acc2, acc3;
#pragma unroll
    for (int i = 0; i < 8; i++) { acc0[i] = bv; acc1[i] = bv; acc2[i] = bv; acc3[i] = bv; }
    const _Float16* a0 = A + (size_t)(tm4 * 64 + l16) * K;
    const _Float16* a1 = a0 + (size_t)16 * K;
    const _Float16* a2 = a0 + (size_t)32 * K;
    const _Float16* a3 = a0 + (size_t)48 * K;
    const _Float16* bp = Bp + (size_t)tn * 512 + lane * 16;
    size_t bstride = (size_t)tilesN * 512;
    for (int k0 = 0; k0 < K; k0 += 32, bp += bstride) {
        v16h b = *(const v16h*)bp;
        __builtin_prefetch(bp + bstride, 0, 1);
        acc0 = WMMA_F16(acc0, afrag(a0, k0, half), b);
        acc1 = WMMA_F16(acc1, afrag(a1, k0, half), b);
        acc2 = WMMA_F16(acc2, afrag(a2, k0, half), b);
        acc3 = WMMA_F16(acc3, afrag(a3, k0, half), b);
    }
#pragma unroll
    for (int r = 0; r < 8; r++) {
        size_t m = tm4 * 64 + r + 8 * half;
        C[(m)*N + col]        = acc0[r];
        C[(m + 16) * N + col] = acc1[r];
        C[(m + 32) * N + col] = acc2[r];
        C[(m + 48) * N + col] = acc3[r];
    }
}

// ---------------------------------------------------------------------------
// QKV GEMM (4 M-tiles/wave) with scatter into attention layouts:
// q -> qh[bh][s][d], k -> kh[bh][s][d], v -> vt[bh][d][s].
// ---------------------------------------------------------------------------
__global__ void gemm_qkv4_kernel(const _Float16* __restrict__ A, const _Float16* __restrict__ Bp,
                                 const float* __restrict__ bias,
                                 _Float16* __restrict__ qh, _Float16* __restrict__ kh,
                                 _Float16* __restrict__ vt, int M, int N, int K) {
    int lane = threadIdx.x & 31;
    int wid  = threadIdx.x >> 5;
    int tilesN = N >> 4;
    int tilesM4 = M >> 6;
    int tile = blockIdx.x * (blockDim.x >> 5) + wid;
    if (tile >= tilesM4 * tilesN) return;
    int tm4 = tile / tilesN, tn = tile - tm4 * tilesN;
    int l16 = lane & 15, half = lane >> 4;
    int col = tn * 16 + l16;
    float bv = bias[col];
    v8f acc0, acc1, acc2, acc3;
#pragma unroll
    for (int i = 0; i < 8; i++) { acc0[i] = bv; acc1[i] = bv; acc2[i] = bv; acc3[i] = bv; }
    const _Float16* a0 = A + (size_t)(tm4 * 64 + l16) * K;
    const _Float16* a1 = a0 + (size_t)16 * K;
    const _Float16* a2 = a0 + (size_t)32 * K;
    const _Float16* a3 = a0 + (size_t)48 * K;
    const _Float16* bp = Bp + (size_t)tn * 512 + lane * 16;
    size_t bstride = (size_t)tilesN * 512;
    for (int k0 = 0; k0 < K; k0 += 32, bp += bstride) {
        v16h b = *(const v16h*)bp;
        __builtin_prefetch(bp + bstride, 0, 1);
        acc0 = WMMA_F16(acc0, afrag(a0, k0, half), b);
        acc1 = WMMA_F16(acc1, afrag(a1, k0, half), b);
        acc2 = WMMA_F16(acc2, afrag(a2, k0, half), b);
        acc3 = WMMA_F16(acc3, afrag(a3, k0, half), b);
    }
    int which = col >> 10;
    int rem   = col & 1023;
    int hh    = rem >> 6;
    int d     = rem & 63;
#pragma unroll
    for (int mt = 0; mt < 4; mt++) {
        v8f acc = (mt == 0) ? acc0 : (mt == 1) ? acc1 : (mt == 2) ? acc2 : acc3;
#pragma unroll
        for (int r = 0; r < 8; r++) {
            int m = tm4 * 64 + mt * 16 + r + 8 * half;
            int b = m >> 9, s = m & 511;
            size_t bh = (size_t)(b * NHc + hh);
            _Float16 v = (_Float16)acc[r];
            if (which == 0)      qh[(bh * Sc + s) * HDc + d] = v;
            else if (which == 1) kh[(bh * Sc + s) * HDc + d] = v;
            else                 vt[(bh * HDc + d) * Sc + s] = v;
        }
    }
}

// ---------------------------------------------------------------------------
// LayerNorm over `cols`, f32 in -> f16 out. One block (256 thr) per row.
// ---------------------------------------------------------------------------
__global__ void ln_kernel(const float* __restrict__ x, const float* __restrict__ g,
                          const float* __restrict__ be, _Float16* __restrict__ y, int cols) {
    __shared__ float red[256];
    __shared__ float stats[2];
    int row = blockIdx.x;
    const float* xr = x + (size_t)row * cols;
    float s = 0.f;
    for (int i = threadIdx.x; i < cols; i += blockDim.x) s += xr[i];
    red[threadIdx.x] = s;
    __syncthreads();
    for (int st = 128; st > 0; st >>= 1) {
        if (threadIdx.x < st) red[threadIdx.x] += red[threadIdx.x + st];
        __syncthreads();
    }
    if (threadIdx.x == 0) stats[0] = red[0] / cols;
    __syncthreads();
    float mean = stats[0];
    float v = 0.f;
    for (int i = threadIdx.x; i < cols; i += blockDim.x) {
        float d = xr[i] - mean;
        v += d * d;
    }
    red[threadIdx.x] = v;
    __syncthreads();
    for (int st = 128; st > 0; st >>= 1) {
        if (threadIdx.x < st) red[threadIdx.x] += red[threadIdx.x + st];
        __syncthreads();
    }
    if (threadIdx.x == 0) stats[1] = rsqrtf(red[0] / cols + 1e-5f);
    __syncthreads();
    float inv = stats[1];
    for (int i = threadIdx.x; i < cols; i += blockDim.x)
        y[(size_t)row * cols + i] = (_Float16)((xr[i] - mean) * inv * g[i] + be[i]);
}

// ---------------------------------------------------------------------------
// Self-attention: one 4-wave workgroup per (b, h). K and V^T staged once into
// LDS (async-to-LDS when available), each wave handles 8 query tiles.
// Dynamic LDS: [4 waves][16][512] f32 scores + K(64KB f16) + Vt(64KB f16)
//              + [4][16] rinv  ==> ~256.25 KB (WGP has 320 KB).
// ---------------------------------------------------------------------------
__global__ void attn_kernel(const _Float16* __restrict__ qh, const _Float16* __restrict__ kh,
                            const _Float16* __restrict__ vt, const float* __restrict__ relemb,
                            const unsigned char* __restrict__ mask,
                            const float* __restrict__ tp, float* __restrict__ out_text) {
    extern __shared__ char smem[];
    int h = blockIdx.x, b = blockIdx.y;
    int tid = threadIdx.x;              // 128
    int lane = tid & 31, wid = tid >> 5;
    int l16 = lane & 15, half = lane >> 4;

    float*    sc   = (float*)(smem + (size_t)wid * 16 * Sc * 4);       // wave-private [16][512]
    _Float16* Kl   = (_Float16*)(smem + (size_t)4 * 16 * Sc * 4);      // [512][64]
    _Float16* Vl   = Kl + (size_t)Sc * HDc;                            // [64][512]
    float*    rinv = (float*)(Vl + (size_t)HDc * Sc) + wid * 16;

    size_t bh = (size_t)(b * NHc + h);
    const _Float16* Qg = qh + bh * Sc * HDc;
    const _Float16* Kg = kh + bh * Sc * HDc;
    const _Float16* Vg = vt + bh * HDc * Sc;

    // ---- stage K and V^T into LDS (32768 f16 each = 4096 b128 chunks) ----
#ifdef HAVE_ASYNC_LDS
    {
        gas_i128* gk = (gas_i128*)Kg;
        las_i128* lk = (las_i128*)Kl;
        gas_i128* gv = (gas_i128*)Vg;
        las_i128* lv = (las_i128*)Vl;
        for (int i = tid; i < (Sc * HDc) / 8; i += 128) {
            __builtin_amdgcn_global_load_async_to_lds_b128(gk + i, lk + i, 0, 0);
            __builtin_amdgcn_global_load_async_to_lds_b128(gv + i, lv + i, 0, 0);
        }
#if __has_builtin(__builtin_amdgcn_s_wait_asynccnt)
        __builtin_amdgcn_s_wait_asynccnt(0);
#else
        asm volatile("s_wait_asynccnt 0x0" ::: "memory");
#endif
    }
#else
    {
        const int4* gk = (const int4*)Kg;
        int4*       lk = (int4*)Kl;
        const int4* gv = (const int4*)Vg;
        int4*       lv = (int4*)Vl;
        for (int i = tid; i < (Sc * HDc) / 8; i += 128) {
            lk[i] = gk[i];
            lv[i] = gv[i];
        }
    }
#endif
    __syncthreads();

    for (int j = 0; j < 8; j++) {
        int qt = wid * 8 + j;
        int q0 = qt * 16;

        // A fragments for Q (16 rows x 64 contraction, two 32-chunks)
        const _Float16* qrow = Qg + (size_t)(q0 + l16) * HDc;
        v16h a_lo = afrag(qrow, 0, half);
        v16h a_hi = afrag(qrow, 32, half);

        // ---- scores = Q K^T * scale + relbias, masked ----
        for (int kt = 0; kt < Sc / 16; kt++) {
            int keyg = kt * 16 + l16;
            const _Float16* krow = Kl + (size_t)keyg * HDc;
            v16h b0 = *(const v16h*)(krow + half * 16);
            v16h b1 = *(const v16h*)(krow + 32 + half * 16);
            v8f acc = {};
            acc = WMMA_F16(acc, a_lo, b0);
            acc = WMMA_F16(acc, a_hi, b1);
            bool msk = mask[b * Sc + keyg] != 0;
#pragma unroll
            for (int r = 0; r < 8; r++) {
                int qg = q0 + r + 8 * half;
                int dd = qg - keyg;
                dd = dd < -128 ? -128 : (dd > 128 ? 128 : dd);
                float v = acc[r] * SCALEc + relemb[(dd + 128) * NHc + h];
                if (!msk) v = -1e9f;
                sc[(r + 8 * half) * Sc + keyg] = v;
            }
        }
        __syncthreads();

        // ---- softmax: lane pair (l16, l16+16) split a row into halves ----
        float* srow = sc + (size_t)l16 * Sc;
        int kbase = half * 256;
        float mx = -1e30f;
        for (int i = kbase; i < kbase + 256; i++) mx = fmaxf(mx, srow[i]);
        mx = fmaxf(mx, __shfl_xor(mx, 16, 32));
        float sum = 0.f;
        for (int i = kbase; i < kbase + 256; i++) {
            float p = __expf(srow[i] - mx);
            srow[i] = p;
            sum += p;
        }
        sum += __shfl_xor(sum, 16, 32);
        if (half == 0) rinv[l16] = 1.f / sum;
        __syncthreads();

        // ---- A*V: 16 x 64 out in 4 n-tiles, contraction over 512 keys ----
        for (int nt = 0; nt < 4; nt++) {
            int d0 = nt * 16;
            v8f acc = {};
            for (int kt = 0; kt < Sc / 32; kt++) {
                int k0 = kt * 32;
                v16h a;
#pragma unroll
                for (int i = 0; i < 8; i++) {
                    a[i]     = (_Float16)sc[(size_t)l16 * Sc + k0 + half * 16 + i];
                    a[i + 8] = (_Float16)sc[(size_t)l16 * Sc + k0 + half * 16 + 8 + i];
                }
                v16h bv = *(const v16h*)(Vl + (size_t)(d0 + l16) * Sc + k0 + half * 16);
                acc = WMMA_F16(acc, a, bv);
            }
            int dg = d0 + l16;
#pragma unroll
            for (int r = 0; r < 8; r++) {
                int qg = q0 + r + 8 * half;
                size_t idx = ((size_t)(b * Sc + qg)) * Hc + h * HDc + dg;
                out_text[idx] = acc[r] * rinv[r + 8 * half] + tp[idx];
            }
        }
        __syncthreads();
    }
}

// ---------------------------------------------------------------------------
// i2t cross-attention (queries are identical broadcast vision token, so the
// post-mean output is a single row per (b,h)). One block (256 thr) per (b,h).
// ---------------------------------------------------------------------------
__global__ void i2t_kernel(const float* __restrict__ vp, const float* __restrict__ ts,
                           const unsigned char* __restrict__ mask, float* __restrict__ out) {
    __shared__ float sc[Sc];
    __shared__ float red[256];
    __shared__ float vq[HDc];
    __shared__ float stats[2];
    int h = blockIdx.x, b = blockIdx.y;
    int t = threadIdx.x;
    if (t < HDc) vq[t] = vp[b * Hc + h * HDc + t];
    __syncthreads();
    for (int k = t; k < Sc; k += 256) {
        const float* tr = ts + ((size_t)(b * Sc + k)) * Hc + h * HDc;
        float s = 0.f;
        for (int d = 0; d < HDc; d++) s += vq[d] * tr[d];
        s *= SCALEc;
        if (!mask[b * Sc + k]) s = -1e9f;
        sc[k] = s;
    }
    __syncthreads();
    red[t] = fmaxf(sc[t], sc[t + 256]);
    __syncthreads();
    for (int st = 128; st > 0; st >>= 1) {
        if (t < st) red[t] = fmaxf(red[t], red[t + st]);
        __syncthreads();
    }
    if (t == 0) stats[0] = red[0];
    __syncthreads();
    float mx = stats[0];
    float e0 = __expf(sc[t] - mx), e1 = __expf(sc[t + 256] - mx);
    sc[t] = e0;
    sc[t + 256] = e1;
    red[t] = e0 + e1;
    __syncthreads();
    for (int st = 128; st > 0; st >>= 1) {
        if (t < st) red[t] += red[t + st];
        __syncthreads();
    }
    if (t == 0) stats[1] = 1.f / red[0];
    __syncthreads();
    float inv = stats[1];
    if (t < HDc) {
        float o = 0.f;
        for (int k = 0; k < Sc; k++)
            o += sc[k] * ts[((size_t)(b * Sc + k)) * Hc + h * HDc + t];
        out[b * Hc + h * HDc + t] = o * inv;
    }
}

// ---------------------------------------------------------------------------
// Final fused adds
// ---------------------------------------------------------------------------
__global__ void final_vision_kernel(const float* __restrict__ vp, const float* __restrict__ gv,
                                    const float* __restrict__ alpha, float* __restrict__ out) {
    int i = blockIdx.x * blockDim.x + threadIdx.x;
    if (i < Bc * Hc) out[i] = vp[i] + alpha[0] * gv[i];
}

__global__ void final_text_kernel(const float* __restrict__ gt, const float* __restrict__ alpha,
                                  float* __restrict__ out_text) {
    size_t i = (size_t)blockIdx.x * blockDim.x + threadIdx.x;
    if (i < (size_t)BSc * Hc) {
        int j = (int)(i & (Hc - 1));
        int b = (int)(i >> 19);   // / (Sc*Hc)
        out_text[i] += alpha[0] * gt[b * Hc + j];
    }
}

// ---------------------------------------------------------------------------
// Host: workspace layout + launch sequence
// ---------------------------------------------------------------------------
extern "C" void kernel_launch(void* const* d_in, const int* in_sizes, int n_in,
                              void* d_out, int out_size, void* d_ws, size_t ws_size,
                              hipStream_t stream) {
    const float* vision = (const float*)d_in[0];
    const float* text   = (const float*)d_in[1];
    const float* W_vp   = (const float*)d_in[2];
    const float* b_vp   = (const float*)d_in[3];
    const float* W_tp   = (const float*)d_in[4];
    const float* b_tp   = (const float*)d_in[5];
    const float* W_qkv  = (const float*)d_in[6];
    const float* b_qkv  = (const float*)d_in[7];
    const float* W_vout = (const float*)d_in[8];
    const float* b_vout = (const float*)d_in[9];
    const float* W_tout = (const float*)d_in[10];
    const float* b_tout = (const float*)d_in[11];
    const float* g_tn   = (const float*)d_in[12];
    const float* be_tn  = (const float*)d_in[13];
    const float* g_i2t  = (const float*)d_in[14];
    const float* be_i2t = (const float*)d_in[15];
    const float* g_t2i  = (const float*)d_in[16];
    const float* be_t2i = (const float*)d_in[17];
    const float* a_i2t  = (const float*)d_in[18];
    const float* a_t2i  = (const float*)d_in[19];
    const float* relemb = (const float*)d_in[20];
    const unsigned char* mask = (const unsigned char*)d_in[21];

    char* ws = (char*)d_ws;
    constexpr size_t SUMS_OFF  = 0;
    constexpr size_t XT_OFF    = 256;
    constexpr size_t XV_OFF    = XT_OFF    + (size_t)BSc * VDc * 2;
    constexpr size_t WQVP_OFF  = XV_OFF    + (size_t)Bc * VDc * 2 + 192;
    constexpr size_t WQTP_OFF  = WQVP_OFF  + (size_t)VDc * Hc * 2;
    constexpr size_t WQQKV_OFF = WQTP_OFF  + (size_t)VDc * Hc * 2;
    constexpr size_t WQVO_OFF  = WQQKV_OFF + (size_t)Hc * 3 * Hc * 2;
    constexpr size_t WQTO_OFF  = WQVO_OFF  + (size_t)Hc * Hc * 2;
    constexpr size_t TP_OFF    = WQTO_OFF  + (size_t)Hc * Hc * 2;
    constexpr size_t VP_OFF    = TP_OFF    + (size_t)BSc * Hc * 4;
    constexpr size_t LN_OFF    = VP_OFF    + (size_t)Bc * Hc * 4;
    constexpr size_t QH_OFF    = LN_OFF    + (size_t)BSc * Hc * 2;
    constexpr size_t KH_OFF    = QH_OFF    + (size_t)BSc * Hc * 2;
    constexpr size_t VT_OFF    = KH_OFF    + (size_t)BSc * Hc * 2;
    constexpr size_t I2T_OFF   = VT_OFF    + (size_t)BSc * Hc * 2;
    constexpr size_t I2TLN_OFF = I2T_OFF   + (size_t)Bc * Hc * 4;
    constexpr size_t T2ILN_OFF = I2TLN_OFF + (size_t)Bc * Hc * 2;
    constexpr size_t GV_OFF    = T2ILN_OFF + (size_t)Bc * Hc * 2;
    constexpr size_t GT_OFF    = GV_OFF    + (size_t)Bc * Hc * 4;

    float*     sums   = (float*)(ws + SUMS_OFF);
    _Float16*  xt_h   = (_Float16*)(ws + XT_OFF);
    _Float16*  xv_h   = (_Float16*)(ws + XV_OFF);
    _Float16*  wq_vp  = (_Float16*)(ws + WQVP_OFF);
    _Float16*  wq_tp  = (_Float16*)(ws + WQTP_OFF);
    _Float16*  wq_qkv = (_Float16*)(ws + WQQKV_OFF);
    _Float16*  wq_vo  = (_Float16*)(ws + WQVO_OFF);
    _Float16*  wq_to  = (_Float16*)(ws + WQTO_OFF);
    float*     tp_f   = (float*)(ws + TP_OFF);
    float*     vp_f   = (float*)(ws + VP_OFF);
    _Float16*  ln_h   = (_Float16*)(ws + LN_OFF);
    _Float16*  qh     = (_Float16*)(ws + QH_OFF);
    _Float16*  kh     = (_Float16*)(ws + KH_OFF);
    _Float16*  vt     = (_Float16*)(ws + VT_OFF);
    float*     i2t_f  = (float*)(ws + I2T_OFF);
    _Float16*  i2tln  = (_Float16*)(ws + I2TLN_OFF);
    _Float16*  t2iln  = (_Float16*)(ws + T2ILN_OFF);
    float*     gv_f   = (float*)(ws + GV_OFF);
    float*     gt_f   = (float*)(ws + GT_OFF);

    float* out_vis  = (float*)d_out;                 // [16, 1024]
    float* out_text = (float*)d_out + Bc * Hc;       // [16, 512, 1024]

    // --- BitNet scales ---
    (void)hipMemsetAsync(sums, 0, 256, stream);
    absmean_kernel<<<512, 256, 0, stream>>>(W_vp,   sums, 0, Hc * VDc);
    absmean_kernel<<<512, 256, 0, stream>>>(W_tp,   sums, 1, Hc * VDc);
    absmean_kernel<<<512, 256, 0, stream>>>(W_qkv,  sums, 2, 3 * Hc * Hc);
    absmean_kernel<<<512, 256, 0, stream>>>(W_vout, sums, 3, Hc * Hc);
    absmean_kernel<<<512, 256, 0, stream>>>(W_tout, sums, 4, Hc * Hc);

    // --- quantize + pack weights into WMMA B-fragment layout ---
    quant_pack_kernel<<<(VDc * Hc + 255) / 256, 256, 0, stream>>>(W_vp,  sums, 0, wq_vp,  VDc, Hc, 1.f / (Hc * VDc));
    quant_pack_kernel<<<(VDc * Hc + 255) / 256, 256, 0, stream>>>(W_tp,  sums, 1, wq_tp,  VDc, Hc, 1.f / (Hc * VDc));
    quant_pack_kernel<<<(3 * Hc * Hc + 255) / 256, 256, 0, stream>>>(W_qkv, sums, 2, wq_qkv, Hc, 3 * Hc, 1.f / (3 * Hc * Hc));
    quant_pack_kernel<<<(Hc * Hc + 255) / 256, 256, 0, stream>>>(W_vout, sums, 3, wq_vo, Hc, Hc, 1.f / (Hc * Hc));
    quant_pack_kernel<<<(Hc * Hc + 255) / 256, 256, 0, stream>>>(W_tout, sums, 4, wq_to, Hc, Hc, 1.f / (Hc * Hc));

    // --- activations to f16 ---
    cvt_h_kernel<<<4096, 256, 0, stream>>>(text,   xt_h, BSc * VDc);
    cvt_h_kernel<<<48,   256, 0, stream>>>(vision, xv_h, Bc * VDc);

    // --- projections ---
    {   // text_proj: [8192,768] x [768,1024], 4 M-tiles per wave
        int tiles = (BSc / 64) * (Hc / 16);
        gemm_bias4_kernel<<<(tiles + 3) / 4, 128, 0, stream>>>(xt_h, wq_tp, b_tp, tp_f, BSc, Hc, VDc);
    }
    {   // vision_proj: [16,768] x [768,1024]
        int tiles = (Bc / 16) * (Hc / 16);
        gemm_bias_kernel<<<(tiles + 3) / 4, 128, 0, stream>>>(xv_h, wq_vp, b_vp, vp_f, Bc, Hc, VDc);
    }

    // --- LN(text_proj) -> f16, then fused QKV GEMM with scatter ---
    ln_kernel<<<BSc, 256, 0, stream>>>(tp_f, g_tn, be_tn, ln_h, Hc);
    {
        int tiles = (BSc / 64) * (3 * Hc / 16);
        gemm_qkv4_kernel<<<(tiles + 3) / 4, 128, 0, stream>>>(ln_h, wq_qkv, b_qkv, qh, kh, vt, BSc, 3 * Hc, Hc);
    }

    // --- self-attention + residual (writes text_self into out_text) ---
    {
        dim3 grid(NHc, Bc);
        size_t smem = (size_t)4 * 16 * Sc * 4      // score buffers
                    + (size_t)Sc * HDc * 2          // K
                    + (size_t)HDc * Sc * 2          // V^T
                    + 4 * 16 * 4;                   // rinv
        attn_kernel<<<grid, 128, smem, stream>>>(qh, kh, vt, relemb, mask, tp_f, out_text);
    }

    // --- i2t branch (collapsed): per-(b,h) attention over text_self ---
    {
        dim3 grid(NHc, Bc);
        i2t_kernel<<<grid, 256, 0, stream>>>(vp_f, out_text, mask, i2t_f);
    }
    ln_kernel<<<Bc, 256, 0, stream>>>(i2t_f, g_i2t, be_i2t, i2tln, Hc);
    // t2i branch collapses to LN(vision_proj)
    ln_kernel<<<Bc, 256, 0, stream>>>(vp_f, g_t2i, be_t2i, t2iln, Hc);

    // --- output bitlinears (16x1024x1024 each) ---
    {
        int tiles = (Bc / 16) * (Hc / 16);
        gemm_bias_kernel<<<(tiles + 3) / 4, 128, 0, stream>>>(i2tln, wq_vo, b_vout, gv_f, Bc, Hc, Hc);
        gemm_bias_kernel<<<(tiles + 3) / 4, 128, 0, stream>>>(t2iln, wq_to, b_tout, gt_f, Bc, Hc, Hc);
    }

    // --- final fused adds ---
    final_vision_kernel<<<(Bc * Hc + 255) / 256, 256, 0, stream>>>(vp_f, gv_f, a_i2t, out_vis);
    final_text_kernel<<<(int)(((size_t)BSc * Hc + 255) / 256), 256, 0, stream>>>(gt_f, a_t2i, out_text);
}
